// CausalSelfAttention_12506944766634
// MI455X (gfx1250) — compile-verified, matrix operand order checked by reading, not run on hardware
//
#include <hip/hip_runtime.h>
#include <math.h>

// ---------------------------------------------------------------------------
// CDNA5 (gfx1250) causal self-attention, bf16 WMMA (16x16x32), f32 accumulate.
// Pipeline: cvt(x) + transpose-cvt(w_qkv,w_proj) -> QKV GEMM -> flash attn -> proj GEMM
// Wave tiles: GEMMs 32(M)x64(N) (8 WMMA / 12 loads per k-step);
// attention 32 queries/wave, 32-key tiles (16 WMMA / 16 global loads per tile).
// ---------------------------------------------------------------------------

typedef __bf16 v16bf __attribute__((ext_vector_type(16)));
typedef float  v8f   __attribute__((ext_vector_type(8)));
typedef unsigned int v4u __attribute__((ext_vector_type(4)));
typedef unsigned int v2u __attribute__((ext_vector_type(2)));

#define BB 4
#define TT 2048
#define CC 1024
#define HH 16
#define DD 64
#define BT (BB*TT)   // 8192 tokens

// round-to-nearest-even f32 -> bf16 (bit pattern)
__device__ __forceinline__ unsigned int f2bf(float f) {
  unsigned int u = __float_as_uint(f);
  u += 0x7FFFu + ((u >> 16) & 1u);
  return u >> 16;
}

union Frag {
  v16bf bf;
  v4u   q[2];
  unsigned short us[16];
};

__device__ __forceinline__ v8f wmma_bf16(v16bf a, v16bf b, v8f c) {
  // (neg_a, A, neg_b, B, c_mod, C, reuse_a, reuse_b)
  return __builtin_amdgcn_wmma_f32_16x16x32_bf16(false, a, false, b, (short)0, c,
                                                 false, false);
}

// --------------------------- conversion kernels ----------------------------

__global__ void cvt_bf16_kernel(const float* __restrict__ src,
                                unsigned short* __restrict__ dst, int n) {
  int i = blockIdx.x * blockDim.x + threadIdx.x;
  int stride = gridDim.x * blockDim.x;
  int n4 = n >> 2;
  const float4* s4 = (const float4*)src;
  v2u* d4 = (v2u*)dst;
  for (; i < n4; i += stride) {
    float4 f = s4[i];
    v2u o;
    o[0] = f2bf(f.x) | (f2bf(f.y) << 16);
    o[1] = f2bf(f.z) | (f2bf(f.w) << 16);
    d4[i] = o;
  }
}

// src[K][N] f32 -> dst[N][K] bf16 (32x32 LDS tile, padded)
__global__ void transpose_cvt_kernel(const float* __restrict__ src,
                                     unsigned short* __restrict__ dst,
                                     int K, int N) {
  __shared__ unsigned short tile[32][33];
  int nb = blockIdx.x * 32;
  int kb = blockIdx.y * 32;
  int tx = threadIdx.x;
  for (int ty = threadIdx.y; ty < 32; ty += 8)
    tile[ty][tx] = (unsigned short)f2bf(src[(size_t)(kb + ty) * N + nb + tx]);
  __syncthreads();
  for (int ty = threadIdx.y; ty < 32; ty += 8)
    dst[(size_t)(nb + ty) * K + kb + tx] = tile[tx][ty];
}

// ------------------------------ QKV GEMM -----------------------------------
// qkv = xb[8192x1024] @ w_qkvT'[1024x3072]; wave tile = 32(M) x 64(N).
// Epilogue scatters: q,k -> [b,h,t,d] bf16 ; v -> [b,h,d,t] bf16 (transposed,
// a contiguous packed 16B store per lane straight from the C-tile layout).
__global__ void __launch_bounds__(256) gemm_qkv_kernel(
    const unsigned short* __restrict__ xb,
    const unsigned short* __restrict__ wT,   // [3072][1024]
    unsigned short* __restrict__ qb,
    unsigned short* __restrict__ kbf,
    unsigned short* __restrict__ vT) {
  const int lane  = threadIdx.x & 31;
  const int wid   = threadIdx.x >> 5;
  const int ln    = lane & 15;
  const int khalf = (lane >> 4) * 8;

  int wtile = blockIdx.x * 8 + wid;          // 12288 wave tiles
  int nIdx = wtile % 48;
  int mIdx = wtile / 48;

  v8f c[2][4] = {};

  const unsigned short* arow0 = xb + (size_t)(mIdx * 32 + ln) * CC;
  const unsigned short* arow1 = arow0 + (size_t)16 * CC;
  const unsigned short* brow[4];
#pragma unroll
  for (int s = 0; s < 4; ++s)
    brow[s] = wT + (size_t)(nIdx * 64 + s * 16 + ln) * CC;

  for (int kk = 0; kk < CC; kk += 32) {
    Frag a0, a1;
    a0.q[0] = *(const v4u*)(arow0 + kk + khalf);
    a0.q[1] = *(const v4u*)(arow0 + kk + 16 + khalf);
    a1.q[0] = *(const v4u*)(arow1 + kk + khalf);
    a1.q[1] = *(const v4u*)(arow1 + kk + 16 + khalf);
#pragma unroll
    for (int s = 0; s < 4; ++s) {
      Frag bf;
      bf.q[0] = *(const v4u*)(brow[s] + kk + khalf);
      bf.q[1] = *(const v4u*)(brow[s] + kk + 16 + khalf);
      c[0][s] = wmma_bf16(a0.bf, bf.bf, c[0][s]);
      c[1][s] = wmma_bf16(a1.bf, bf.bf, c[1][s]);
    }
  }

#pragma unroll
  for (int mi = 0; mi < 2; ++mi) {
    const int btbase = mIdx * 32 + mi * 16 + khalf;  // 8-aligned run: one batch
    const int bidx = btbase >> 11;
    const int t0   = btbase & (TT - 1);
#pragma unroll
    for (int s = 0; s < 4; ++s) {
      int col = nIdx * 64 + s * 16 + ln;
      int seg = col >> 10;                   // 0=q 1=k 2=v (uniform per s)
      int c1  = col & (CC - 1);
      int h   = c1 >> 6;
      int d   = c1 & 63;
      if (seg == 2) {
        v4u packed;
#pragma unroll
        for (int i = 0; i < 4; ++i) {
          unsigned lo = f2bf(c[mi][s][2 * i]);
          unsigned hi = f2bf(c[mi][s][2 * i + 1]);
          packed[i] = lo | (hi << 16);
        }
        *(v4u*)(vT + ((size_t)((bidx * HH + h) * DD + d)) * TT + t0) = packed;
      } else {
        unsigned short* dst = (seg == 1) ? kbf : qb;
        size_t base = ((size_t)((bidx * HH + h) * TT + t0)) * DD + d;
#pragma unroll
        for (int j = 0; j < 8; ++j)
          dst[base + (size_t)j * DD] = (unsigned short)f2bf(c[mi][s][j]);
      }
    }
  }
}

// ---------------------------- flash attention ------------------------------
// One wave = 32 queries (two 16-row tiles) of one (b,h). Key tiles of 32.
// q0 % 32 == 0 => every processed key tile has >=1 valid key per row, so the
// online softmax needs no fully-masked-row guard. Per key tile: 8 WMMA for
// S = Q K^T (shared K frags), softmax with shfl_xor 16-lane row reductions,
// P -> LDS bf16 bounce (s_wait_dscnt 0), 8 WMMA for O += P V (shared V frags).
__global__ void __launch_bounds__(128) attn_kernel(
    const unsigned short* __restrict__ qb,
    const unsigned short* __restrict__ kbf,
    const unsigned short* __restrict__ vT,
    unsigned short* __restrict__ yb) {
  __shared__ __align__(16) unsigned short pbuf[4 * 2 * 16 * 32];  // 8 KB

  const int lane  = threadIdx.x & 31;
  const int wid   = threadIdx.x >> 5;
  const int ln    = lane & 15;
  const int khalf = (lane >> 4) * 8;

  int wtile = blockIdx.x * 4 + wid;          // 4096 wave tiles
  int qt = wtile & 63;                       // T/32 = 64
  int bh = wtile >> 6;
  int h = bh & (HH - 1);
  int b = bh >> 4;
  int q0 = qt * 32;

  const unsigned short* qbase = qb  + (size_t)(b * HH + h) * TT * DD;
  const unsigned short* kbase = kbf + (size_t)(b * HH + h) * TT * DD;
  const unsigned short* vbase = vT  + (size_t)(b * HH + h) * DD * TT;

  Frag qf[2][2];
#pragma unroll
  for (int r = 0; r < 2; ++r)
#pragma unroll
    for (int dc = 0; dc < 2; ++dc) {
      const unsigned short* p =
          qbase + (size_t)(q0 + r * 16 + ln) * DD + dc * 32 + khalf;
      qf[r][dc].q[0] = *(const v4u*)(p);
      qf[r][dc].q[1] = *(const v4u*)(p + 16);
    }

  v8f acc[2][4] = {};
  float m_i[2][8], l_i[2][8];
#pragma unroll
  for (int r = 0; r < 2; ++r)
#pragma unroll
    for (int j = 0; j < 8; ++j) { m_i[r][j] = -__builtin_inff(); l_i[r][j] = 0.0f; }

  const float sc = 0.125f * 1.44269504088896340736f;  // 1/sqrt(64) * log2(e)
  unsigned short* pw = pbuf + wid * (2 * 16 * 32);

  const int ktmax = q0 >> 5;
  for (int kt = 0; kt <= ktmax; ++kt) {
    int key0 = kt * 32 + ln;
    int key1 = key0 + 16;

    v8f s[2][2] = {};
#pragma unroll
    for (int dc = 0; dc < 2; ++dc) {
      Frag kf0, kf1;
      const unsigned short* r0 = kbase + (size_t)key0 * DD + dc * 32 + khalf;
      const unsigned short* r1 = kbase + (size_t)key1 * DD + dc * 32 + khalf;
      kf0.q[0] = *(const v4u*)(r0);
      kf0.q[1] = *(const v4u*)(r0 + 16);
      kf1.q[0] = *(const v4u*)(r1);
      kf1.q[1] = *(const v4u*)(r1 + 16);
      s[0][0] = wmma_bf16(qf[0][dc].bf, kf0.bf, s[0][0]);
      s[0][1] = wmma_bf16(qf[0][dc].bf, kf1.bf, s[0][1]);
      s[1][0] = wmma_bf16(qf[1][dc].bf, kf0.bf, s[1][0]);
      s[1][1] = wmma_bf16(qf[1][dc].bf, kf1.bf, s[1][1]);
    }

#pragma unroll
    for (int r = 0; r < 2; ++r) {
      const int qrow0 = q0 + r * 16 + khalf;  // query row for j=0, this lane
#pragma unroll
      for (int j = 0; j < 8; ++j) {
        int qg = qrow0 + j;
        float t0v = (key0 <= qg) ? s[r][0][j] * sc : -__builtin_inff();
        float t1v = (key1 <= qg) ? s[r][1][j] * sc : -__builtin_inff();
        float mr = fmaxf(t0v, t1v);
#pragma unroll
        for (int off = 1; off < 16; off <<= 1)
          mr = fmaxf(mr, __shfl_xor(mr, off, 32));
        float newm = fmaxf(m_i[r][j], mr);
        float sf = exp2f(m_i[r][j] - newm);
        float p0 = exp2f(t0v - newm);
        float p1 = exp2f(t1v - newm);
        float ls = p0 + p1;
#pragma unroll
        for (int off = 1; off < 16; off <<= 1)
          ls += __shfl_xor(ls, off, 32);
        m_i[r][j] = newm;
        l_i[r][j] = l_i[r][j] * sf + ls;
#pragma unroll
        for (int dt = 0; dt < 4; ++dt) acc[r][dt][j] *= sf;
        int m = khalf + j;
        pw[r * 512 + m * 32 + ln]      = (unsigned short)f2bf(p0);
        pw[r * 512 + m * 32 + 16 + ln] = (unsigned short)f2bf(p1);
      }
    }

    // order LDS stores before the fragment re-load (CDNA5 split counter)
    asm volatile("s_wait_dscnt 0" ::: "memory");

    Frag pf[2];
#pragma unroll
    for (int r = 0; r < 2; ++r) {
      const unsigned short* pr = pw + r * 512 + ln * 32 + khalf;
      pf[r].q[0] = *(const v4u*)(pr);
      pf[r].q[1] = *(const v4u*)(pr + 16);
    }

#pragma unroll
    for (int dt = 0; dt < 4; ++dt) {
      Frag vf;
      const unsigned short* vr =
          vbase + (size_t)(dt * 16 + ln) * TT + kt * 32 + khalf;
      vf.q[0] = *(const v4u*)(vr);
      vf.q[1] = *(const v4u*)(vr + 16);
      acc[0][dt] = wmma_bf16(pf[0].bf, vf.bf, acc[0][dt]);
      acc[1][dt] = wmma_bf16(pf[1].bf, vf.bf, acc[1][dt]);
    }
  }

#pragma unroll
  for (int r = 0; r < 2; ++r) {
    float inv[8];
#pragma unroll
    for (int j = 0; j < 8; ++j) inv[j] = 1.0f / l_i[r][j];
#pragma unroll
    for (int dt = 0; dt < 4; ++dt) {
      int col = h * DD + dt * 16 + ln;
#pragma unroll
      for (int j = 0; j < 8; ++j) {
        int t = q0 + r * 16 + khalf + j;
        yb[((size_t)(b * TT + t)) * CC + col] =
            (unsigned short)f2bf(acc[r][dt][j] * inv[j]);
      }
    }
  }
}

// ------------------------------ proj GEMM ----------------------------------
// out = yb[8192x1024] @ w_projT'[1024x1024]; wave tile = 32(M) x 64(N), f32 out.
__global__ void __launch_bounds__(256) gemm_proj_kernel(
    const unsigned short* __restrict__ yb,
    const unsigned short* __restrict__ wT,   // [1024][1024]
    float* __restrict__ out) {
  const int lane  = threadIdx.x & 31;
  const int wid   = threadIdx.x >> 5;
  const int ln    = lane & 15;
  const int khalf = (lane >> 4) * 8;

  int wtile = blockIdx.x * 8 + wid;          // 4096 wave tiles
  int nIdx = wtile & 15;
  int mIdx = wtile >> 4;

  v8f c[2][4] = {};

  const unsigned short* arow0 = yb + (size_t)(mIdx * 32 + ln) * CC;
  const unsigned short* arow1 = arow0 + (size_t)16 * CC;
  const unsigned short* brow[4];
#pragma unroll
  for (int s = 0; s < 4; ++s)
    brow[s] = wT + (size_t)(nIdx * 64 + s * 16 + ln) * CC;

  for (int kk = 0; kk < CC; kk += 32) {
    Frag a0, a1;
    a0.q[0] = *(const v4u*)(arow0 + kk + khalf);
    a0.q[1] = *(const v4u*)(arow0 + kk + 16 + khalf);
    a1.q[0] = *(const v4u*)(arow1 + kk + khalf);
    a1.q[1] = *(const v4u*)(arow1 + kk + 16 + khalf);
#pragma unroll
    for (int s = 0; s < 4; ++s) {
      Frag bf;
      bf.q[0] = *(const v4u*)(brow[s] + kk + khalf);
      bf.q[1] = *(const v4u*)(brow[s] + kk + 16 + khalf);
      c[0][s] = wmma_bf16(a0.bf, bf.bf, c[0][s]);
      c[1][s] = wmma_bf16(a1.bf, bf.bf, c[1][s]);
    }
  }

#pragma unroll
  for (int mi = 0; mi < 2; ++mi) {
    const int btbase = mIdx * 32 + mi * 16 + khalf;
#pragma unroll
    for (int s = 0; s < 4; ++s) {
      int col = nIdx * 64 + s * 16 + ln;
#pragma unroll
      for (int j = 0; j < 8; ++j)
        out[(size_t)(btbase + j) * CC + col] = c[mi][s][j];
    }
  }
}

// ------------------------------- launcher ----------------------------------
extern "C" void kernel_launch(void* const* d_in, const int* in_sizes, int n_in,
                              void* d_out, int out_size, void* d_ws, size_t ws_size,
                              hipStream_t stream) {
  (void)in_sizes; (void)n_in; (void)out_size; (void)ws_size;
  const float* x      = (const float*)d_in[0];
  const float* w_qkv  = (const float*)d_in[1];
  const float* w_proj = (const float*)d_in[2];
  float* out = (float*)d_out;

  char* ws = (char*)d_ws;
  size_t off = 0;
  unsigned short* xb     = (unsigned short*)(ws + off); off += (size_t)BT * CC * 2;      // 16 MB
  unsigned short* wqkvT  = (unsigned short*)(ws + off); off += (size_t)3 * CC * CC * 2;  //  6 MB
  unsigned short* wprojT = (unsigned short*)(ws + off); off += (size_t)CC * CC * 2;      //  2 MB
  unsigned short* qb     = (unsigned short*)(ws + off); off += (size_t)BT * CC * 2;      // 16 MB
  unsigned short* kb     = (unsigned short*)(ws + off); off += (size_t)BT * CC * 2;      // 16 MB
  unsigned short* vT     = (unsigned short*)(ws + off); off += (size_t)BT * CC * 2;      // 16 MB
  unsigned short* yb     = (unsigned short*)(ws + off); off += (size_t)BT * CC * 2;      // 16 MB

  cvt_bf16_kernel<<<1024, 256, 0, stream>>>(x, xb, BT * CC);
  transpose_cvt_kernel<<<dim3(96, 32), dim3(32, 8), 0, stream>>>(w_qkv, wqkvT, CC, 3 * CC);
  transpose_cvt_kernel<<<dim3(32, 32), dim3(32, 8), 0, stream>>>(w_proj, wprojT, CC, CC);
  gemm_qkv_kernel<<<1536, 256, 0, stream>>>(xb, wqkvT, qb, kb, vT);
  attn_kernel<<<1024, 128, 0, stream>>>(qb, kb, vT, yb);
  gemm_proj_kernel<<<512, 256, 0, stream>>>(yb, wprojT, out);
}
